// NSAAttention_90400471646451
// MI455X (gfx1250) — matrix operand order repaired and checked by reading, not compiled, not run on hardware
//
#include <hip/hip_runtime.h>
#include <hip/hip_bf16.h>
#include <math.h>

// ---------------- static problem config (matches reference) ----------------
#define SLEN 2048
#define DIMM 1024
#define NH   16
#define NG   4
#define HPG  4
#define DKD  64
#define DVD  64
#define LCMP 32
#define DSTR 16
#define LSEL 64
#define NSEL 16
#define WWIN 512
#define NCMP 127     // (S - L)/D + 1
#define NJ   32      // S / LSEL
#define GHH  32
#define NEGF (-1e30f)

typedef __attribute__((ext_vector_type(16))) __bf16 v16bf;
typedef __attribute__((ext_vector_type(8)))  __bf16 v8bf;
typedef __attribute__((ext_vector_type(8)))  float  v8f;

__device__ __forceinline__ float wave_max(float v) {
  for (int o = 16; o; o >>= 1) v = fmaxf(v, __shfl_xor(v, o, 32));
  return v;
}
__device__ __forceinline__ float wave_sum(float v) {
  for (int o = 16; o; o >>= 1) v += __shfl_xor(v, o, 32);
  return v;
}

// ---------------- f32 -> bf16 convert ---------------------------------------
__global__ void cvt_bf16_kernel(const float* __restrict__ src, __bf16* __restrict__ dst,
                                int n) {
  int i = blockIdx.x * blockDim.x + threadIdx.x;
  if (i < n) dst[i] = (__bf16)src[i];
}

// ---------------- f32 [K,N] -> bf16 transposed [N,K] ------------------------
__global__ void cvtT_bf16_kernel(const float* __restrict__ W, __bf16* __restrict__ Wt,
                                 int K, int N) {
  int i = blockIdx.x * blockDim.x + threadIdx.x;
  if (i >= K * N) return;
  int n = i / K, k = i % K;
  Wt[i] = (__bf16)W[(size_t)k * N + n];
}

// ---------------------------------------------------------------------------
// WMMA bf16 GEMM: C[M,N] f32 = A[M,K] bf16 row-major  @  Bt[N,K] bf16 (B^T)
// One wave computes a 16x64 C strip: one A fragment reused across 4 B frags,
// 4 v_wmma_f32_16x16x32_bf16 per K-step of 32.
// Fragment layouts used here (contiguous k, same for A and B; not AMD's table order):
//   A: lane m=lane&15, hi=lane>>4; elem e -> K=k0+16*hi+e (two contig 16B loads)
//   B: lane n=lane&15, hi=lane>>4; elem e -> K=k0+16*hi+e  (contig 32B in Bt)
//   C/D: N=lane&15; vgpr v -> M = 8*(lane>>4)+v
// ---------------------------------------------------------------------------
template <int KD, int ND>
__global__ void gemm_bf16t_wmma(const __bf16* __restrict__ A,
                                const __bf16* __restrict__ Bt,
                                float* __restrict__ C, int M) {
  constexpr int TN = ND / 64;                 // 64-wide N strips
  const int wid = blockIdx.x * (blockDim.x >> 5) + (threadIdx.x >> 5);
  const int tilesM = M >> 4;
  if (wid >= tilesM * TN) return;
  const int tm   = wid / TN;
  const int tn4  = wid % TN;
  const int lane = threadIdx.x & 31;
  const int mr   = lane & 15;
  const int hi   = lane >> 4;

  v8f acc0 = {}, acc1 = {}, acc2 = {}, acc3 = {};
  const __bf16* arow = A + (size_t)(tm * 16 + mr) * KD;
  const __bf16* b0r  = Bt + (size_t)(tn4 * 64 +  0 + mr) * KD;
  const __bf16* b1r  = Bt + (size_t)(tn4 * 64 + 16 + mr) * KD;
  const __bf16* b2r  = Bt + (size_t)(tn4 * 64 + 32 + mr) * KD;
  const __bf16* b3r  = Bt + (size_t)(tn4 * 64 + 48 + mr) * KD;

#pragma unroll 2
  for (int k0 = 0; k0 < KD; k0 += 32) {
    __builtin_prefetch(arow + k0 + 128, 0, 1);   // global_prefetch path
    v8bf alo = *(const v8bf*)(arow + k0 + 16 * hi);
    v8bf ahi = *(const v8bf*)(arow + k0 + 16 * hi + 8);
    v16bf a = __builtin_shufflevector(alo, ahi,
        0, 1, 2, 3, 4, 5, 6, 7, 8, 9, 10, 11, 12, 13, 14, 15);
    v16bf b0 = *(const v16bf*)(b0r + k0 + 16 * hi);
    v16bf b1 = *(const v16bf*)(b1r + k0 + 16 * hi);
    v16bf b2 = *(const v16bf*)(b2r + k0 + 16 * hi);
    v16bf b3 = *(const v16bf*)(b3r + k0 + 16 * hi);
    acc0 = __builtin_amdgcn_wmma_f32_16x16x32_bf16(false, a, false, b0, (short)0, acc0, false, false);
    acc1 = __builtin_amdgcn_wmma_f32_16x16x32_bf16(false, a, false, b1, (short)0, acc1, false, false);
    acc2 = __builtin_amdgcn_wmma_f32_16x16x32_bf16(false, a, false, b2, (short)0, acc2, false, false);
    acc3 = __builtin_amdgcn_wmma_f32_16x16x32_bf16(false, a, false, b3, (short)0, acc3, false, false);
  }

  float* crow = C + (size_t)(tm * 16 + hi * 8) * ND + tn4 * 64 + mr;
#pragma unroll
  for (int v = 0; v < 8; ++v) {
    crow[(size_t)v * ND +  0] = acc0[v];
    crow[(size_t)v * ND + 16] = acc1[v];
    crow[(size_t)v * ND + 32] = acc2[v];
    crow[(size_t)v * ND + 48] = acc3[v];
  }
}

// ---------------- RoPE (llama half-rotation), in place: buf[S, nh, 64] ------
__global__ void rope_kernel(float* __restrict__ buf, int nh) {
  int idx = blockIdx.x * blockDim.x + threadIdx.x;
  int total = SLEN * nh * 32;
  if (idx >= total) return;
  int i = idx & 31;
  int h = (idx >> 5) % nh;
  int s = idx / (32 * nh);
  float* p = buf + ((size_t)s * nh + h) * 64;
  double inv = pow(10000.0, -(double)i / 32.0);
  double ang = (double)s * inv;
  float c = (float)cos(ang), sn = (float)sin(ang);
  float x1 = p[i], x2 = p[i + 32];
  p[i]      = x1 * c - x2 * sn;
  p[i + 32] = x2 * c + x1 * sn;
}

// ---------------- avg-pool compressed K/V (len L=32, stride D=16) -----------
__global__ void pool_kernel(const float* __restrict__ Kc, const float* __restrict__ Vc,
                            float* __restrict__ Kp, float* __restrict__ Vp) {
  int idx = blockIdx.x * blockDim.x + threadIdx.x;
  if (idx >= NG * NCMP * DKD) return;
  int d = idx & 63;
  int c = (idx >> 6) % NCMP;
  int g = idx / (64 * NCMP);
  float ak = 0.f, av = 0.f;
  for (int l = 0; l < LCMP; ++l) {
    int u = c * DSTR + l;
    ak += Kc[((size_t)u * NG + g) * DKD + d];
    av += Vc[((size_t)u * NG + g) * DKD + d];
  }
  Kp[((size_t)g * NCMP + c) * DKD + d] = ak * (1.f / LCMP);
  Vp[((size_t)g * NCMP + c) * DKD + d] = av * (1.f / LCMP);
}

// ---------------- compressed attention + selection-block scores -------------
__global__ void cmp_attn_kernel(const float* __restrict__ Q, const float* __restrict__ Kp,
                                const float* __restrict__ Vp, float* __restrict__ Ocmp,
                                float* __restrict__ pgrp) {
  int sg = blockIdx.x;
  int s = sg / NG, g = sg % NG;
  int wave = threadIdx.x >> 5, lane = threadIdx.x & 31;
  int h = g * HPG + wave;
  __shared__ float qs[HPG][DKD];
  __shared__ float pb[HPG][128];
  __shared__ float sinv[HPG];
  qs[wave][lane]      = Q[((size_t)s * NH + h) * DKD + lane];
  qs[wave][lane + 32] = Q[((size_t)s * NH + h) * DKD + lane + 32];
  __syncthreads();

  float sc[4];
  bool  vis[4];
  float mx = NEGF;
#pragma unroll
  for (int t = 0; t < 4; ++t) {
    int c = lane + 32 * t;
    vis[t] = (c < NCMP) && (c * DSTR + LCMP <= s + 1);
    float v = NEGF;
    if (vis[t]) {
      const float* kr = Kp + ((size_t)g * NCMP + c) * DKD;
      float a = 0.f;
      for (int d = 0; d < DKD; ++d) a += qs[wave][d] * kr[d];
      v = a * 0.125f;
    }
    sc[t] = v;
    mx = fmaxf(mx, v);
  }
  mx = wave_max(mx);
  float sum = 0.f;
#pragma unroll
  for (int t = 0; t < 4; ++t) {
    int c = lane + 32 * t;
    float e = vis[t] ? __expf(sc[t] - mx) : 0.f;
    pb[wave][c] = e;
    sum += e;
  }
  sum = wave_sum(sum);
  float inv = (sum > 0.f) ? 1.f / sum : 0.f;   // empty row (s<31) -> zeros
  if (lane == 0) sinv[wave] = inv;
  __syncthreads();

#pragma unroll
  for (int dd = 0; dd < 2; ++dd) {
    int d = lane + 32 * dd;
    float o = 0.f;
    for (int c = 0; c < NCMP; ++c)
      o += pb[wave][c] * Vp[((size_t)g * NCMP + c) * DVD + d];
    Ocmp[((size_t)s * NH + h) * DVD + d] = o * inv;
  }

  if (wave == 0) {
    int j = lane;
    float acc = 0.f;
    int c0 = 4 * j - 1; if (c0 < 0) c0 = 0;
    int c1 = 4 * j + 3; if (c1 > NCMP - 1) c1 = NCMP - 1;
    for (int c = c0; c <= c1; ++c) {
      float lo  = fmaxf((float)(c * DSTR), (float)(j * LSEL));
      float hi2 = fminf((float)(c * DSTR + LCMP), (float)(j * LSEL + LSEL));
      float ov  = fmaxf(0.f, hi2 - lo) * (1.f / LCMP);
      if (ov > 0.f) {
        float ph = pb[0][c] * sinv[0] + pb[1][c] * sinv[1]
                 + pb[2][c] * sinv[2] + pb[3][c] * sinv[3];
        acc += ph * ov;
      }
    }
    pgrp[(size_t)sg * NJ + j] = acc;
  }
}

// ---------------- top-NSEL selection blocks -> bitmask ----------------------
__global__ void select_kernel(const float* __restrict__ pgrp,
                              unsigned int* __restrict__ blk) {
  int sg = blockIdx.x * blockDim.x + threadIdx.x;
  if (sg >= SLEN * NG) return;
  int s = sg / NG;
  int cur = s / LSEL;
  float sc[NJ];
#pragma unroll
  for (int j = 0; j < NJ; ++j) {
    float v = pgrp[(size_t)sg * NJ + j];
    if (j * LSEL > s) v = NEGF;              // visibility
    if (j == 0 || j == cur) v = 1e9f;        // forced blocks
    sc[j] = v;
  }
  unsigned int mask = 0;
  for (int n = 0; n < NSEL; ++n) {           // top-k, ties -> lowest index
    int bj = 0; float bv = -3e38f;
    for (int j = 0; j < NJ; ++j)
      if (!((mask >> j) & 1u) && sc[j] > bv) { bv = sc[j]; bj = j; }
    mask |= (1u << bj);
  }
  blk[sg] = mask;
}

// ---------------- selected-block attention ----------------------------------
__global__ void sel_attn_kernel(const float* __restrict__ Q, const float* __restrict__ Ks,
                                const float* __restrict__ Vs,
                                const unsigned int* __restrict__ blk,
                                float* __restrict__ Osel) {
  int sg = blockIdx.x;
  int s = sg / NG, g = sg % NG;
  int wave = threadIdx.x >> 5, lane = threadIdx.x & 31;
  int h = g * HPG + wave;
  __shared__ float qs[HPG][DKD];
  __shared__ float pb[HPG][NSEL * LSEL];
  __shared__ int   jl[NSEL];
  __shared__ int   njs;
  if (threadIdx.x == 0) {
    unsigned int m = blk[sg];
    int n = 0;
    for (int j = 0; j < NJ && n < NSEL; ++j)
      if ((m >> j) & 1u) jl[n++] = j;
    njs = n;
  }
  qs[wave][lane]      = Q[((size_t)s * NH + h) * DKD + lane];
  qs[wave][lane + 32] = Q[((size_t)s * NH + h) * DKD + lane + 32];
  __syncthreads();
  int nj = njs;

  float mx = NEGF;
  for (int bi = 0; bi < nj; ++bi) {
    int j = jl[bi];
#pragma unroll
    for (int uu = 0; uu < 2; ++uu) {
      int u = j * LSEL + lane + 32 * uu;
      float v = NEGF;
      if (u <= s) {                          // causal within selected block
        const float* kr = Ks + ((size_t)u * NG + g) * DKD;
        float a = 0.f;
        for (int d = 0; d < DKD; ++d) a += qs[wave][d] * kr[d];
        v = a * 0.125f;
      }
      pb[wave][bi * LSEL + lane + 32 * uu] = v;
      mx = fmaxf(mx, v);
    }
  }
  mx = wave_max(mx);
  float sum = 0.f;
  for (int bi = 0; bi < nj; ++bi) {
#pragma unroll
    for (int uu = 0; uu < 2; ++uu) {
      int idx = bi * LSEL + lane + 32 * uu;
      float v = pb[wave][idx];
      float e = (v > -1e29f) ? __expf(v - mx) : 0.f;
      pb[wave][idx] = e;
      sum += e;
    }
  }
  sum = wave_sum(sum);
  float inv = (sum > 0.f) ? 1.f / sum : 0.f;
  __syncthreads();

#pragma unroll
  for (int dd = 0; dd < 2; ++dd) {
    int d = lane + 32 * dd;
    float o = 0.f;
    for (int bi = 0; bi < nj; ++bi) {
      int j = jl[bi];
      for (int t = 0; t < LSEL; ++t)
        o += pb[wave][bi * LSEL + t] * Vs[((size_t)(j * LSEL + t) * NG + g) * DVD + d];
    }
    Osel[((size_t)s * NH + h) * DVD + d] = o * inv;
  }
}

// ---------------- sliding-window attention ----------------------------------
__global__ void win_attn_kernel(const float* __restrict__ Q, const float* __restrict__ Kw,
                                const float* __restrict__ Vw, float* __restrict__ Owin) {
  int sg = blockIdx.x;
  int s = sg / NG, g = sg % NG;
  int wave = threadIdx.x >> 5, lane = threadIdx.x & 31;
  int h = g * HPG + wave;
  __shared__ float qs[HPG][DKD];
  __shared__ float pb[HPG][WWIN];
  qs[wave][lane]      = Q[((size_t)s * NH + h) * DKD + lane];
  qs[wave][lane + 32] = Q[((size_t)s * NH + h) * DKD + lane + 32];
  __syncthreads();
  int u0 = s - (WWIN - 1); if (u0 < 0) u0 = 0;
  int cnt = s - u0 + 1;

  float mx = NEGF;
  for (int t = 0; t < WWIN / 32; ++t) {
    int ks = lane + 32 * t;
    float v = NEGF;
    if (ks < cnt) {
      const float* kr = Kw + ((size_t)(u0 + ks) * NG + g) * DKD;
      float a = 0.f;
      for (int d = 0; d < DKD; ++d) a += qs[wave][d] * kr[d];
      v = a * 0.125f;
    }
    pb[wave][ks] = v;
    mx = fmaxf(mx, v);
  }
  mx = wave_max(mx);
  float sum = 0.f;
  for (int t = 0; t < WWIN / 32; ++t) {
    int ks = lane + 32 * t;
    float v = pb[wave][ks];
    float e = (v > -1e29f) ? __expf(v - mx) : 0.f;
    pb[wave][ks] = e;
    sum += e;
  }
  sum = wave_sum(sum);
  float inv = 1.f / sum;    // window always contains u = s
  __syncthreads();

#pragma unroll
  for (int dd = 0; dd < 2; ++dd) {
    int d = lane + 32 * dd;
    float o = 0.f;
    for (int ks = 0; ks < cnt; ++ks)
      o += pb[wave][ks] * Vw[((size_t)(u0 + ks) * NG + g) * DVD + d];
    Owin[((size_t)s * NH + h) * DVD + d] = o * inv;
  }
}

// ---------------- gating MLP (one wave per (s,g); GH == 32 == wave) ---------
__global__ void gate_kernel(const float* __restrict__ Q, const float* __restrict__ w1,
                            const float* __restrict__ b1, const float* __restrict__ w2,
                            const float* __restrict__ b2, float* __restrict__ gates) {
  int sg = blockIdx.x;
  int s = sg / NG, g = sg % NG;
  int lane = threadIdx.x & 31;
  __shared__ float qm[DKD];
#pragma unroll
  for (int dd = 0; dd < 2; ++dd) {
    int d = lane + 32 * dd;
    float a = 0.f;
    for (int hp = 0; hp < HPG; ++hp)
      a += Q[((size_t)s * NH + g * HPG + hp) * DKD + d];
    qm[d] = a * 0.25f;
  }
  __syncthreads();
  float a = b1[lane];
  for (int d = 0; d < DKD; ++d) a += qm[d] * w1[d * GHH + lane];
  float h1 = a / (1.f + __expf(-a));        // silu
  float c0 = h1 * w2[lane * 3 + 0];
  float c1 = h1 * w2[lane * 3 + 1];
  float c2 = h1 * w2[lane * 3 + 2];
  c0 = wave_sum(c0); c1 = wave_sum(c1); c2 = wave_sum(c2);
  if (lane == 0) {
    float l0 = c0 + b2[0], l1 = c1 + b2[1], l2 = c2 + b2[2];
    float m = fmaxf(l0, fmaxf(l1, l2));
    float e0 = __expf(l0 - m), e1 = __expf(l1 - m), e2 = __expf(l2 - m);
    float ss = e0 + e1 + e2;
    float p0 = e0 / ss, p1 = e1 / ss, p2 = e2 / ss;
    int amax = 0; float bm = l0;
    if (l1 > bm) { bm = l1; amax = 1; }
    if (l2 > bm) { bm = l2; amax = 2; }
    float t2 = -3e38f;
    if (amax != 0) t2 = fmaxf(t2, l0);
    if (amax != 1) t2 = fmaxf(t2, l1);
    if (amax != 2) t2 = fmaxf(t2, l2);
    if (bm - t2 > 50.f) {
      p0 = (amax == 0) ? 1.f : 0.f;
      p1 = (amax == 1) ? 1.f : 0.f;
      p2 = (amax == 2) ? 1.f : 0.f;
    }
    gates[(size_t)sg * 3 + 0] = p0;
    gates[(size_t)sg * 3 + 1] = p1;
    gates[(size_t)sg * 3 + 2] = p2;
  }
}

// ---------------- gated combine ---------------------------------------------
__global__ void combine_kernel(const float* __restrict__ Ocmp, const float* __restrict__ Osel,
                               const float* __restrict__ Owin, const float* __restrict__ gates,
                               float* __restrict__ Oc) {
  size_t idx = (size_t)blockIdx.x * blockDim.x + threadIdx.x;
  if (idx >= (size_t)SLEN * NH * DVD) return;
  int h = (int)((idx / DVD) % NH);
  int s = (int)(idx / (NH * DVD));
  int g = h / HPG;
  const float* gt = gates + ((size_t)s * NG + g) * 3;
  Oc[idx] = gt[0] * Ocmp[idx] + gt[1] * Osel[idx] + gt[2] * Owin[idx];
}

// ---------------------------------------------------------------------------
extern "C" void kernel_launch(void* const* d_in, const int* in_sizes, int n_in,
                              void* d_out, int out_size, void* d_ws, size_t ws_size,
                              hipStream_t stream) {
  const float* x     = (const float*)d_in[0];
  const float* W_Q   = (const float*)d_in[1];
  const float* W_Ks  = (const float*)d_in[2];
  const float* W_Vs  = (const float*)d_in[3];
  const float* W_Kw  = (const float*)d_in[4];
  const float* W_Vw  = (const float*)d_in[5];
  const float* W_Kc  = (const float*)d_in[6];
  const float* W_Vc  = (const float*)d_in[7];
  const float* W_out = (const float*)d_in[8];
  const float* gw1   = (const float*)d_in[9];
  const float* gb1   = (const float*)d_in[10];
  const float* gw2   = (const float*)d_in[11];
  const float* gb2   = (const float*)d_in[12];

  float* w = (float*)d_ws;
  float* Q     = w; w += (size_t)SLEN * NH * DKD;     // f32 buffers
  float* Ksel  = w; w += (size_t)SLEN * NG * DKD;
  float* Vsel  = w; w += (size_t)SLEN * NG * DVD;
  float* Kwin  = w; w += (size_t)SLEN * NG * DKD;
  float* Vwin  = w; w += (size_t)SLEN * NG * DVD;
  float* Kcmp  = w; w += (size_t)SLEN * NG * DKD;
  float* Vcmp  = w; w += (size_t)SLEN * NG * DVD;
  float* KcmpP = w; w += (size_t)NG * NCMP * DKD;
  float* VcmpP = w; w += (size_t)NG * NCMP * DVD;
  float* Ocmp  = w; w += (size_t)SLEN * NH * DVD;
  float* Osel  = w; w += (size_t)SLEN * NH * DVD;
  float* Owin  = w; w += (size_t)SLEN * NH * DVD;
  float* Ocomb = w; w += (size_t)SLEN * NH * DVD;
  float* pgrp  = w; w += (size_t)SLEN * NG * NJ;
  float* gates = w; w += (size_t)SLEN * NG * 3;
  unsigned int* blk = (unsigned int*)w; w += (size_t)SLEN * NG;  // 1 u32 each
  // bf16 operand buffers (sizes counted in float units = 2 bf16 per unit)
  __bf16* xbf   = (__bf16*)w; w += (size_t)SLEN * DIMM / 2;
  __bf16* WQt   = (__bf16*)w; w += (size_t)DIMM * DIMM / 2;
  __bf16* WKst  = (__bf16*)w; w += (size_t)DIMM * NG * DKD / 2;
  __bf16* WVst  = (__bf16*)w; w += (size_t)DIMM * NG * DVD / 2;
  __bf16* WKwt  = (__bf16*)w; w += (size_t)DIMM * NG * DKD / 2;
  __bf16* WVwt  = (__bf16*)w; w += (size_t)DIMM * NG * DVD / 2;
  __bf16* WKct  = (__bf16*)w; w += (size_t)DIMM * NG * DKD / 2;
  __bf16* WVct  = (__bf16*)w; w += (size_t)DIMM * NG * DVD / 2;
  __bf16* WOt   = (__bf16*)w; w += (size_t)DIMM * DIMM / 2;
  __bf16* Ocbf  = (__bf16*)w; w += (size_t)SLEN * NH * DVD / 2;

  float* out = (float*)d_out;

  // --- operand conversion / weight transpose (bf16) ---
  cvt_bf16_kernel<<<(SLEN * DIMM + 255) / 256, 256, 0, stream>>>(x, xbf, SLEN * DIMM);
  cvtT_bf16_kernel<<<(DIMM * DIMM + 255) / 256, 256, 0, stream>>>(W_Q, WQt, DIMM, DIMM);
  {
    int n = DIMM * NG * DKD;
    int grid = (n + 255) / 256;
    cvtT_bf16_kernel<<<grid, 256, 0, stream>>>(W_Ks, WKst, DIMM, NG * DKD);
    cvtT_bf16_kernel<<<grid, 256, 0, stream>>>(W_Vs, WVst, DIMM, NG * DVD);
    cvtT_bf16_kernel<<<grid, 256, 0, stream>>>(W_Kw, WKwt, DIMM, NG * DKD);
    cvtT_bf16_kernel<<<grid, 256, 0, stream>>>(W_Vw, WVwt, DIMM, NG * DVD);
    cvtT_bf16_kernel<<<grid, 256, 0, stream>>>(W_Kc, WKct, DIMM, NG * DKD);
    cvtT_bf16_kernel<<<grid, 256, 0, stream>>>(W_Vc, WVct, DIMM, NG * DVD);
  }
  cvtT_bf16_kernel<<<(DIMM * DIMM + 255) / 256, 256, 0, stream>>>(W_out, WOt, DIMM, DIMM);

  // --- projections via bf16 WMMA GEMM (16x64 strip per wave, 4 wmma/K-step) ---
  {
    int waves = (SLEN / 16) * (DIMM / 64);      // 2048
    gemm_bf16t_wmma<DIMM, DIMM><<<waves / 8, 256, 0, stream>>>(xbf, WQt, Q, SLEN);
  }
  {
    int waves = (SLEN / 16) * ((NG * DKD) / 64);  // 512
    int grid = waves / 8;
    gemm_bf16t_wmma<DIMM, NG * DKD><<<grid, 256, 0, stream>>>(xbf, WKst, Ksel, SLEN);
    gemm_bf16t_wmma<DIMM, NG * DVD><<<grid, 256, 0, stream>>>(xbf, WVst, Vsel, SLEN);
    gemm_bf16t_wmma<DIMM, NG * DKD><<<grid, 256, 0, stream>>>(xbf, WKwt, Kwin, SLEN);
    gemm_bf16t_wmma<DIMM, NG * DVD><<<grid, 256, 0, stream>>>(xbf, WVwt, Vwin, SLEN);
    gemm_bf16t_wmma<DIMM, NG * DKD><<<grid, 256, 0, stream>>>(xbf, WKct, Kcmp, SLEN);
    gemm_bf16t_wmma<DIMM, NG * DVD><<<grid, 256, 0, stream>>>(xbf, WVct, Vcmp, SLEN);
  }

  // --- RoPE on Q and all K branches (K_cmp roped BEFORE pooling, as in ref) ---
  rope_kernel<<<(SLEN * NH * 32 + 255) / 256, 256, 0, stream>>>(Q, NH);
  rope_kernel<<<(SLEN * NG * 32 + 255) / 256, 256, 0, stream>>>(Ksel, NG);
  rope_kernel<<<(SLEN * NG * 32 + 255) / 256, 256, 0, stream>>>(Kwin, NG);
  rope_kernel<<<(SLEN * NG * 32 + 255) / 256, 256, 0, stream>>>(Kcmp, NG);

  // --- compressed pooling ---
  pool_kernel<<<(NG * NCMP * DKD + 255) / 256, 256, 0, stream>>>(Kcmp, Vcmp, KcmpP, VcmpP);

  // --- branch attentions ---
  cmp_attn_kernel<<<SLEN * NG, 128, 0, stream>>>(Q, KcmpP, VcmpP, Ocmp, pgrp);
  select_kernel<<<(SLEN * NG + 255) / 256, 256, 0, stream>>>(pgrp, blk);
  sel_attn_kernel<<<SLEN * NG, 128, 0, stream>>>(Q, Ksel, Vsel, blk, Osel);
  win_attn_kernel<<<SLEN * NG, 128, 0, stream>>>(Q, Kwin, Vwin, Owin);

  // --- gating + combine ---
  gate_kernel<<<SLEN * NG, 32, 0, stream>>>(Q, gw1, gb1, gw2, gb2, gates);
  combine_kernel<<<(SLEN * NH * DVD + 255) / 256, 256, 0, stream>>>(Ocmp, Osel, Owin, gates, Ocomb);

  // --- output projection (WMMA) into d_out ---
  cvt_bf16_kernel<<<(SLEN * NH * DVD + 255) / 256, 256, 0, stream>>>(Ocomb, Ocbf, SLEN * NH * DVD);
  {
    int waves = (SLEN / 16) * (DIMM / 64);
    gemm_bf16t_wmma<DIMM, DIMM><<<waves / 8, 256, 0, stream>>>(Ocbf, WOt, out, SLEN);
  }
}
